// SelfAttention_5746666242285
// MI455X (gfx1250) — matrix operand order repaired and checked by reading, not compiled, not run on hardware
//
#include <hip/hip_runtime.h>
#include <stdint.h>

// ---- CDNA5 (gfx1250) wave32 WMMA attention ----
typedef __attribute__((ext_vector_type(16))) __bf16 v16bf;
typedef __attribute__((ext_vector_type(8)))  float  v8f;
typedef __attribute__((ext_vector_type(4)))  unsigned int u32x4;
typedef __attribute__((ext_vector_type(8)))  int    i32x8;
typedef __attribute__((ext_vector_type(4)))  int    i32x4;

union BF16x16 {
  v16bf v;
  __bf16 h[16];
  u32x4  q[2];
};
union BF16x8 {
  u32x4  q;
  __bf16 h[8];
};

#if __has_builtin(__builtin_amdgcn_tensor_load_to_lds)
#define USE_TDM 1
#else
#define USE_TDM 0
#endif

__device__ __forceinline__ v8f wmma_bf16(v16bf a, v16bf b, v8f c) {
  // D(16x16 f32) = A(16x32 bf16) * B(32x16 bf16) + C
  return __builtin_amdgcn_wmma_f32_16x16x32_bf16(false, a, false, b, (short)0, c,
                                                 false, false);
}

// ============================================================================
// Kernel 0: one-time fp32 -> bf16 conversion (x and the three weights).
// Removes all cvt VALU from the GEMM hot loops and halves fragment bytes.
// ============================================================================
__global__ __launch_bounds__(256) void cvt_bf16_kernel(
    const float* __restrict__ src, __bf16* __restrict__ dst, int n8)
{
  const int i = blockIdx.x * 256 + threadIdx.x;   // 8 elements per thread
  if (i >= n8) return;
  const float* p = src + (size_t)i * 8;
  BF16x8 o;
#pragma unroll
  for (int j = 0; j < 8; ++j) o.h[j] = (__bf16)p[j];
  *((u32x4*)dst + i) = o.q;
}

// ============================================================================
// Kernel 1: QKV projection.  One wave per 16x64 output slice (4 d-tiles),
// reusing each A fragment across 4 WMMAs.
//   q[b][t][d] = (x . Wq^T) * 1/sqrt(768)  (bf16, pre-scaled)
//   k[b][t][d] =  x . Wk^T                 (bf16)
//   vt[b][d][t] = x . Wv^T                 (bf16, transposed for PV B-operand)
// ============================================================================
__global__ __launch_bounds__(256) void qkv_proj_kernel(
    const __bf16* __restrict__ xb,  const __bf16* __restrict__ Wqb,
    const __bf16* __restrict__ Wkb, const __bf16* __restrict__ Wvb,
    __bf16* __restrict__ qws, __bf16* __restrict__ kws, __bf16* __restrict__ vtws)
{
  const int lane = threadIdx.x & 31;
  const int wv   = threadIdx.x >> 5;
  int gid = blockIdx.x * 8 + wv;            // 36864 slices total
  const int mat = gid / 12288;              // 16 * 64 * 12
  int rem = gid - mat * 12288;
  const int b  = rem / 768;                 // 64 * 12
  rem -= b * 768;
  const int t0 = (rem / 12) * 16;
  const int d0 = (rem % 12) * 64;

  const int hf = lane >> 4;
  const int ln = lane & 15;

  const __bf16* W    = (mat == 0) ? Wqb : (mat == 1) ? Wkb : Wvb;
  const __bf16* arow = xb + ((size_t)b * 1024 + t0 + ln) * 768;   // A row M=ln

  v8f zeroV = {};
  v8f acc[4];
#pragma unroll
  for (int j = 0; j < 4; ++j) acc[j] = zeroV;

  for (int kc = 0; kc < 768; kc += 32) {
    BF16x16 a;
    const u32x4* ap = (const u32x4*)(arow + kc);
    a.q[0] = ap[2 * hf];                     // K = kc + half*16 .. +7
    a.q[1] = ap[2 * hf + 1];                 // K = kc + half*16 + 8 .. +15
#pragma unroll
    for (int j = 0; j < 4; ++j) {
      const u32x4* bp =
          (const u32x4*)(W + (size_t)(d0 + j * 16 + ln) * 768 + kc + hf * 16);
      BF16x16 bb;
      bb.q[0] = bp[0]; bb.q[1] = bp[1];      // K = kc + half*16 .. +15
      acc[j] = wmma_bf16(a.v, bb.v, acc[j]);
    }
  }

  // D layout: element i -> (M = i + 8*half, N = ln)
  if (mat == 0) {
#pragma unroll
    for (int j = 0; j < 4; ++j)
#pragma unroll
      for (int i = 0; i < 8; ++i) {
        const int m = i + 8 * hf;
        qws[((size_t)b * 1024 + t0 + m) * 768 + d0 + j * 16 + ln] =
            (__bf16)(acc[j][i] * 0.0360843918243516f);   // 1/sqrt(768)
      }
  } else if (mat == 1) {
#pragma unroll
    for (int j = 0; j < 4; ++j)
#pragma unroll
      for (int i = 0; i < 8; ++i) {
        const int m = i + 8 * hf;
        kws[((size_t)b * 1024 + t0 + m) * 768 + d0 + j * 16 + ln] = (__bf16)acc[j][i];
      }
  } else {
#pragma unroll
    for (int j = 0; j < 4; ++j)
#pragma unroll
      for (int i = 0; i < 8; ++i) {
        const int m = i + 8 * hf;
        vtws[((size_t)b * 768 + d0 + j * 16 + ln) * 1024 + t0 + m] = (__bf16)acc[j][i];
      }
  }
}

// ============================================================================
// Kernel 2: flash attention.  One workgroup (4 waves) per 16-query tile.
// Q tile DMA'd into LDS via the Tensor Data Mover; per 32-key step the waves
// split the K=768 score reduction, combine partials in LDS, run the online
// softmax redundantly (16-lane butterflies), stage P through LDS, and each
// wave accumulates a 192-wide d-slice of O (12 f32 WMMA tiles).
// ============================================================================
__global__ __launch_bounds__(128) void attn_kernel(
    const __bf16* __restrict__ qws, const __bf16* __restrict__ kws,
    const __bf16* __restrict__ vtws, float* __restrict__ out)
{
  __shared__ __align__(16) __bf16 Qs[16 * 768];        // 24 KB
  __shared__ float  Sp[4][2][16][16];                  // 8 KB partial scores
  __shared__ __align__(16) __bf16 Ps[16 * 32];         // 1 KB probabilities

  const int tid  = threadIdx.x;
  const int lane = tid & 31;
  const int wv   = tid >> 5;
  const int hf   = lane >> 4;
  const int ln   = lane & 15;
  const int b    = blockIdx.x >> 6;          // 64 query tiles per batch
  const int t0   = (blockIdx.x & 63) * 16;
  const int dbase = wv * 192;                // this wave's output d-slice

#if USE_TDM
  if (wv == 0) {
    // Tensor DMA descriptor (CDNA5 ISA): 2D tensor,
    // data_size=2B, tile 768x16, row stride 768 elements.
    const unsigned long long ga =
        (unsigned long long)(uintptr_t)(qws + ((size_t)b * 1024 + t0) * 768);
    const unsigned int lds = (unsigned int)(uintptr_t)(void*)Qs;  // LDS byte offs
    u32x4 g0;
    g0[0] = 1u;                                        // count=1 user D#
    g0[1] = lds;                                       // lds_addr
    g0[2] = (unsigned int)(ga & 0xffffffffu);          // global_addr[31:0]
    g0[3] = (unsigned int)((ga >> 32) & 0x1ffffffu)    // global_addr[56:32]
            | (2u << 30);                              // type = 2 ("image")
    i32x8 g1;
    g1[0] = (1 << 16);       // workgroup_mask=0, data_size=1 (2 bytes)
    g1[1] = (768 << 16);     // tensor_dim0[15:0] = 768
    g1[2] = (16 << 16);      // tensor_dim0 hi = 0, tensor_dim1 = 16
    g1[3] = (768 << 16);     // tensor_dim1 hi = 0, tile_dim0 = 768
    g1[4] = 16;              // tile_dim1 = 16, tile_dim2 = 0
    g1[5] = 768;             // tensor_dim0_stride[31:0] = 768
    g1[6] = 0;               // stride hi / tensor_dim1_stride lo
    g1[7] = 0;
    i32x4 gz4 = {0, 0, 0, 0};                          // D# groups 2/3 (<=2D)
    i32x8 gz8 = {0, 0, 0, 0, 0, 0, 0, 0};
    // 6-arg toolchain variant: (g0, g1, g2, g3, g_extra, cpol)
    __builtin_amdgcn_tensor_load_to_lds(g0, g1, gz4, gz4, gz8, 0);
    __builtin_amdgcn_s_wait_tensorcnt(0);
  }
#else
  {          // cooperative 24KB copy: q tile is contiguous in ws
    const u32x4* src = (const u32x4*)(qws + ((size_t)b * 1024 + t0) * 768);
    u32x4* dst = (u32x4*)Qs;
#pragma unroll
    for (int i = 0; i < 12; ++i) dst[i * 128 + tid] = src[i * 128 + tid];
  }
#endif
  __syncthreads();

  float mrow[8], lrow[8];
  v8f acc[12];
  v8f zeroV = {};
#pragma unroll
  for (int i = 0; i < 8; ++i) { mrow[i] = -3.0e38f; lrow[i] = 0.f; }
#pragma unroll
  for (int t = 0; t < 12; ++t) acc[t] = zeroV;

  const int nsteps = t0 / 32 + 1;            // causal: keys s < t0+16
  for (int it = 0; it < nsteps; ++it) {
    const int s0 = it * 32;

    // prefetch next step's K/V stream (global_prefetch_b8)
    if (it + 1 < nsteps) {
      const int sn = s0 + 32;
      __builtin_prefetch(kws + ((size_t)b * 1024 + sn + ln) * 768 + dbase, 0, 1);
      __builtin_prefetch(kws + ((size_t)b * 1024 + sn + 16 + ln) * 768 + dbase, 0, 1);
      __builtin_prefetch(vtws + ((size_t)b * 768 + dbase + ln) * 1024 + sn, 0, 1);
    }

    // ---- partial S = Q[:, krange] . K^T[krange, s0..s0+31], krange per wave
    v8f s0acc = zeroV, s1acc = zeroV;
#pragma unroll
    for (int k6 = 0; k6 < 6; ++k6) {
      const int kc = dbase + k6 * 32;
      BF16x16 a, b0, b1;
      const u32x4* qa = (const u32x4*)(Qs + ln * 768 + kc);
      a.q[0] = qa[2 * hf];                   // K = kc + half*16
      a.q[1] = qa[2 * hf + 1];               // K = kc + half*16 + 8
      const u32x4* kp0 =
          (const u32x4*)(kws + ((size_t)b * 1024 + s0 + ln) * 768 + kc + hf * 16);
      const u32x4* kp1 =
          (const u32x4*)(kws + ((size_t)b * 1024 + s0 + 16 + ln) * 768 + kc + hf * 16);
      b0.q[0] = kp0[0]; b0.q[1] = kp0[1];
      b1.q[0] = kp1[0]; b1.q[1] = kp1[1];
      s0acc = wmma_bf16(a.v, b0.v, s0acc);
      s1acc = wmma_bf16(a.v, b1.v, s1acc);
    }
#pragma unroll
    for (int i = 0; i < 8; ++i) {
      const int m = i + 8 * hf;
      Sp[wv][0][m][ln] = s0acc[i];
      Sp[wv][1][m][ln] = s1acc[i];
    }
    __syncthreads();   // partials visible; prev-iter P reads done

    // ---- combine partials + online softmax (all waves redundantly)
    v8f svec;
#pragma unroll
    for (int i = 0; i < 8; ++i) {
      const int m = i + 8 * hf;
      float v0 = Sp[0][0][m][ln] + Sp[1][0][m][ln] + Sp[2][0][m][ln] + Sp[3][0][m][ln];
      float v1 = Sp[0][1][m][ln] + Sp[1][1][m][ln] + Sp[2][1][m][ln] + Sp[3][1][m][ln];
      const int trow = t0 + m;
      v0 = (s0 + ln      <= trow) ? v0 : -3.0e38f;   // causal mask
      v1 = (s0 + 16 + ln <= trow) ? v1 : -3.0e38f;
      float rmax = fmaxf(v0, v1);                    // row lives in one 16-lane half
      rmax = fmaxf(rmax, __shfl_xor(rmax, 1));
      rmax = fmaxf(rmax, __shfl_xor(rmax, 2));
      rmax = fmaxf(rmax, __shfl_xor(rmax, 4));
      rmax = fmaxf(rmax, __shfl_xor(rmax, 8));
      const float mnew = fmaxf(mrow[i], rmax);
      const float p0 = __expf(v0 - mnew);
      const float p1 = __expf(v1 - mnew);
      float rsum = p0 + p1;
      rsum += __shfl_xor(rsum, 1);
      rsum += __shfl_xor(rsum, 2);
      rsum += __shfl_xor(rsum, 4);
      rsum += __shfl_xor(rsum, 8);
      const float sc = __expf(mrow[i] - mnew);
      lrow[i] = lrow[i] * sc + rsum;
      mrow[i] = mnew;
      svec[i] = sc;
      if (wv == 0) {                          // reshape D-layout -> row-major LDS
        Ps[m * 32 + ln]      = (__bf16)p0;
        Ps[m * 32 + 16 + ln] = (__bf16)p1;
      }
    }
    __syncthreads();   // P ready; safe to overwrite Sp next iter

    // ---- O = O*scale + P(16x32) . V(32x16) for this wave's 12 d-tiles
    BF16x16 pa;
    const u32x4* pp = (const u32x4*)(Ps + ln * 32);
    pa.q[0] = pp[2 * hf];
    pa.q[1] = pp[2 * hf + 1];
#pragma unroll
    for (int t = 0; t < 12; ++t) {
      const int dcol = dbase + t * 16 + ln;
      const u32x4* vp =
          (const u32x4*)(vtws + ((size_t)b * 768 + dcol) * 1024 + s0 + hf * 16);
      BF16x16 vb;
      vb.q[0] = vp[0]; vb.q[1] = vp[1];
      acc[t] = acc[t] * svec;
      acc[t] = wmma_bf16(pa.v, vb.v, acc[t]);
    }
  }

  // ---- epilogue: out = acc / l
#pragma unroll
  for (int t = 0; t < 12; ++t) {
#pragma unroll
    for (int i = 0; i < 8; ++i) {
      const int m = i + 8 * hf;
      out[((size_t)b * 1024 + t0 + m) * 768 + dbase + t * 16 + ln] =
          acc[t][i] / lrow[i];
    }
  }
}

extern "C" void kernel_launch(void* const* d_in, const int* in_sizes, int n_in,
                              void* d_out, int out_size, void* d_ws, size_t ws_size,
                              hipStream_t stream) {
  (void)in_sizes; (void)n_in; (void)out_size; (void)ws_size;
  const float* x  = (const float*)d_in[0];
  const float* Wq = (const float*)d_in[1];
  const float* Wk = (const float*)d_in[2];
  const float* Wv = (const float*)d_in[3];
  float* out = (float*)d_out;

  // workspace layout (bf16 elements):
  //   q (24MB) | k (24MB) | v^T (24MB) | x_bf16 (24MB) | Wq/Wk/Wv bf16 (3x1.125MB)
  const size_t NXT = (size_t)16 * 1024 * 768;   // 12,582,912
  const size_t NW  = (size_t)768 * 768;         // 589,824
  __bf16* qws = (__bf16*)d_ws;
  __bf16* kws = qws + NXT;
  __bf16* vt  = kws + NXT;
  __bf16* xb  = vt  + NXT;
  __bf16* wqb = xb  + NXT;
  __bf16* wkb = wqb + NW;
  __bf16* wvb = wkb + NW;

  // one-time fp32 -> bf16 conversion passes
  cvt_bf16_kernel<<<(int)(NXT / 8 / 256), 256, 0, stream>>>(x, xb, (int)(NXT / 8));
  cvt_bf16_kernel<<<(int)(NW / 8 / 256), 256, 0, stream>>>(Wq, wqb, (int)(NW / 8));
  cvt_bf16_kernel<<<(int)(NW / 8 / 256), 256, 0, stream>>>(Wk, wkb, (int)(NW / 8));
  cvt_bf16_kernel<<<(int)(NW / 8 / 256), 256, 0, stream>>>(Wv, wvb, (int)(NW / 8));

  // 16 b * 64 t-tiles * 12 d-slices * 3 matrices = 36864 waves / 8 per block
  qkv_proj_kernel<<<4608, 256, 0, stream>>>(xb, wqb, wkb, wvb, qws, kws, vt);
  attn_kernel<<<1024, 128, 0, stream>>>(qws, kws, vt, out);
}
